// GatedDeltaNetMixer_40733469835716
// MI455X (gfx1250) — matrix operand order repaired and checked by reading, not compiled, not run on hardware
//
#include <hip/hip_runtime.h>
#include <hip/hip_bf16.h>

// ---------------------------------------------------------------------------
// Gated DeltaNet mixer for MI455X (gfx1250), wave32 + WMMA bf16.
//   0) one-shot f32->bf16 conversion of x, W_in, W_o (bf16-resident operands)
//   1) qkvz = x @ W_in          (WMMA bf16 GEMM, async-LDS A tiles)
//   2) ba   = x @ W_ba          (small dot-product kernel)
//   3) causal depthwise conv(4) + SiLU + l2norm(q,k) + beta/g gates
//   4) gated delta-rule recurrence (state in VGPRs, 1 block per (b,h))
//   5) gated RMSNorm * silu(z)  -> y emitted directly in bf16
//   6) out  = y @ W_o           (WMMA bf16 GEMM)
// ---------------------------------------------------------------------------

typedef __attribute__((ext_vector_type(16))) __bf16 v16bf;
typedef __attribute__((ext_vector_type(8)))  __bf16 v8bf;
typedef __attribute__((ext_vector_type(4)))  __bf16 v4bf;
typedef __attribute__((ext_vector_type(2)))  __bf16 v2bf;
typedef __attribute__((ext_vector_type(8)))  float  v8f;
typedef __attribute__((ext_vector_type(4)))  int    v4i;

constexpr int Bz = 2, Tz = 4096, Dz = 1024, Hz = 16, HDz = 64, KVz = 1024;

#if __has_builtin(__builtin_amdgcn_global_load_async_to_lds_b128) && \
    __has_builtin(__builtin_amdgcn_s_wait_asynccnt)
#define GDN_ASYNC_LDS 1
typedef __attribute__((address_space(1))) v4i* gas_v4i;   // global int4*
typedef __attribute__((address_space(3))) v4i* las_v4i;   // LDS int4*
#else
#define GDN_ASYNC_LDS 0
#endif

// ---------------------------------------------------------------------------
// f32 -> bf16 streaming conversion (n % 1024 == 0)
// ---------------------------------------------------------------------------
__global__ __launch_bounds__(256)
void gdn_cvt_bf16(const float* __restrict__ in, __bf16* __restrict__ out) {
  const size_t i = (size_t)blockIdx.x * blockDim.x + threadIdx.x;
  float4 f = ((const float4*)in)[i];
  v4bf o;
  o[0] = (__bf16)f.x; o[1] = (__bf16)f.y; o[2] = (__bf16)f.z; o[3] = (__bf16)f.w;
  ((v4bf*)out)[i] = o;
}

// ---------------------------------------------------------------------------
// WMMA GEMM: C[M,N] = A[M,K] * B[K,N]; A,B bf16 row-major, C f32.
// Block tile 128x128, 4 waves (2x2), wave tile 64x64 (4x4 wmma 16x16x32).
// A tile staged via GLOBAL_LOAD_ASYNC_TO_LDS_B128 (ASYNCcnt); B tile staged
// through VGPRs performing the transpose so both fragments assemble as
// 2 x ds_load_b128.  M,N % 128 == 0, K % 32 == 0.
// ---------------------------------------------------------------------------
__global__ __launch_bounds__(128)
void gdn_wmma_gemm_bf16(const __bf16* __restrict__ A, const __bf16* __restrict__ Bm,
                        float* __restrict__ C, int M, int N, int K) {
  __shared__ __bf16 sA[128][40];    // 128 rows x 32 K, padded (stride 80B)
  __shared__ __bf16 sBt[128][40];   // 128 cols x 32 K, padded (B transposed)

  const int tid  = threadIdx.x;
  const int lane = tid & 31;
  const int wave = tid >> 5;
  const int wm   = (wave >> 1) * 64;
  const int wn   = (wave &  1) * 64;
  const int half = lane >> 4;
  const int l15  = lane & 15;

  const int blockM = blockIdx.x * 128;
  const int blockN = blockIdx.y * 128;

  // B staging decomposition: thread -> (k-pair, 16-col chunk)
  const int kp = tid >> 3;          // 0..15  -> K rows {2kp, 2kp+1}
  const int c0 = (tid & 7) * 16;    // column chunk start

  v8f acc[4][4];
#pragma unroll
  for (int i = 0; i < 4; ++i)
#pragma unroll
    for (int j = 0; j < 4; ++j)
      acc[i][j] = (v8f){0.f, 0.f, 0.f, 0.f, 0.f, 0.f, 0.f, 0.f};

  for (int k0 = 0; k0 < K; k0 += 32) {
    // --- stage A tile: one row per thread, 32 bf16 = 4 x b128 ---
    {
      const __bf16* src = A + (size_t)(blockM + tid) * K + k0;
#if GDN_ASYNC_LDS
#pragma unroll
      for (int i = 0; i < 4; ++i)
        __builtin_amdgcn_global_load_async_to_lds_b128(
            (gas_v4i)(v4i*)(src + i * 8),
            (las_v4i)(v4i*)&sA[tid][i * 8],
            0, 0);
#else
      v8bf t0 = ((const v8bf*)src)[0];
      v8bf t1 = ((const v8bf*)src)[1];
      v8bf t2 = ((const v8bf*)src)[2];
      v8bf t3 = ((const v8bf*)src)[3];
      *(v8bf*)&sA[tid][0]  = t0;
      *(v8bf*)&sA[tid][8]  = t1;
      *(v8bf*)&sA[tid][16] = t2;
      *(v8bf*)&sA[tid][24] = t3;
#endif
    }
    // --- stage B tile transposed: rows 2kp,2kp+1, cols c0..c0+15 ---
    {
      const __bf16* r0 = Bm + (size_t)(k0 + 2 * kp) * N + blockN + c0;
      const __bf16* r1 = r0 + N;
      v8bf a0 = ((const v8bf*)r0)[0], a1 = ((const v8bf*)r0)[1];
      v8bf b0 = ((const v8bf*)r1)[0], b1 = ((const v8bf*)r1)[1];
      if (k0 + 32 < K)
        __builtin_prefetch((const char*)r0 + (size_t)32 * N * sizeof(__bf16), 0, 0);
#pragma unroll
      for (int c = 0; c < 8; ++c) {
        v2bf p; p[0] = a0[c]; p[1] = b0[c];
        *(v2bf*)&sBt[c0 + c][2 * kp] = p;
        v2bf q; q[0] = a1[c]; q[1] = b1[c];
        *(v2bf*)&sBt[c0 + 8 + c][2 * kp] = q;
      }
    }
#if GDN_ASYNC_LDS
    __builtin_amdgcn_s_wait_asynccnt(0);
#endif
    __syncthreads();

    // --- assemble fragments (ISA 7.12.2 layouts), 2 x b128 each ---
    v16bf afrag[4], bfrag[4];
#pragma unroll
    for (int ti = 0; ti < 4; ++ti) {
      const int row = wm + ti * 16 + l15;
      v8bf lo = *(const v8bf*)&sA[row][half * 8];        // K = half*8 + 0..7
      v8bf hi = *(const v8bf*)&sA[row][16 + half * 8];   // K = 16+half*8 + 0..7
      afrag[ti] = __builtin_shufflevector(lo, hi, 0, 1, 2, 3, 4, 5, 6, 7,
                                          8, 9, 10, 11, 12, 13, 14, 15);
    }
#pragma unroll
    for (int tj = 0; tj < 4; ++tj) {
      const int col = wn + tj * 16 + l15;
      v8bf lo = *(const v8bf*)&sBt[col][half * 8];       // K = half*8 + 0..7
      v8bf hi = *(const v8bf*)&sBt[col][16 + half * 8];  // K = 16+half*8 + 0..7
      bfrag[tj] = __builtin_shufflevector(lo, hi, 0, 1, 2, 3, 4, 5, 6, 7,
                                          8, 9, 10, 11, 12, 13, 14, 15);
    }

#pragma unroll
    for (int ti = 0; ti < 4; ++ti)
#pragma unroll
      for (int tj = 0; tj < 4; ++tj)
        acc[ti][tj] = __builtin_amdgcn_wmma_f32_16x16x32_bf16(
            false, afrag[ti], false, bfrag[tj], (short)0, acc[ti][tj],
            false, false);
    __syncthreads();
  }

  // --- writeback: D layout, M = vgpr + half*8, N = lane&15 ---
#pragma unroll
  for (int ti = 0; ti < 4; ++ti)
#pragma unroll
    for (int i = 0; i < 8; ++i) {
      const size_t row = (size_t)(blockM + wm + ti * 16 + half * 8 + i);
      float* crow = C + row * N + blockN + wn + l15;
#pragma unroll
      for (int tj = 0; tj < 4; ++tj)
        crow[tj * 16] = acc[ti][tj][i];
    }
}

// ---------------------------------------------------------------------------
// ba = x @ W_ba : [BT,1024] @ [1024,32]
// ---------------------------------------------------------------------------
__global__ __launch_bounds__(256)
void gdn_ba_gemm(const float* __restrict__ x, const float* __restrict__ W_ba,
                 float* __restrict__ ba) {
  const int idx   = blockIdx.x * blockDim.x + threadIdx.x;
  const int col   = idx & 31;
  const int rowbt = idx >> 5;
  const float* xr = x + (size_t)rowbt * Dz;
  float s = 0.f;
  for (int kk = 0; kk < Dz; ++kk) s += xr[kk] * W_ba[kk * 32 + col];
  ba[idx] = s;
}

// ---------------------------------------------------------------------------
// causal depthwise conv(k=4, pad 3) + SiLU over concat(q,k,v), then
// l2norm(q)*HD^-0.5, l2norm(k); beta = sigmoid(b), g = -A*softplus(a+dt_bias).
// ---------------------------------------------------------------------------
__device__ __forceinline__ float silu_f(float x) {
  return x / (1.f + __expf(-x));
}

__global__ __launch_bounds__(64)
void gdn_conv_prep(const float* __restrict__ qkvz, const float* __restrict__ ba,
                   const float* __restrict__ conv_w,
                   const float* __restrict__ dt_bias,
                   const float* __restrict__ A_log,
                   float* __restrict__ qp, float* __restrict__ kp,
                   float* __restrict__ vp, float* __restrict__ g,
                   float* __restrict__ beta) {
  const int bid = blockIdx.x;           // b*T*H + t*H + h
  const int h = bid % Hz;
  const int t = (bid / Hz) % Tz;
  const int b = bid / (Hz * Tz);
  const int d = threadIdx.x;

  __shared__ float red[HDz];

  float vals[3];
#pragma unroll
  for (int s = 0; s < 3; ++s) {
    const int col = s * KVz + h * HDz + d;
    const float* w = conv_w + (size_t)col * 4;
    float acc = 0.f;
#pragma unroll
    for (int j = 0; j < 4; ++j) {
      const int tt = t - 3 + j;
      if (tt >= 0)
        acc += w[j] * qkvz[((size_t)b * Tz + tt) * (4 * KVz) + col];
    }
    vals[s] = silu_f(acc);
  }

  red[d] = vals[0] * vals[0];
  __syncthreads();
  for (int off = 32; off > 0; off >>= 1) {
    if (d < off) red[d] += red[d + off];
    __syncthreads();
  }
  const float rq = rsqrtf(red[0] + 1e-6f) * 0.125f;  // * HD^-0.5
  __syncthreads();

  red[d] = vals[1] * vals[1];
  __syncthreads();
  for (int off = 32; off > 0; off >>= 1) {
    if (d < off) red[d] += red[d + off];
    __syncthreads();
  }
  const float rk = rsqrtf(red[0] + 1e-6f);

  const size_t out = ((size_t)b * Tz + t) * KVz + h * HDz + d;
  qp[out] = vals[0] * rq;
  kp[out] = vals[1] * rk;
  vp[out] = vals[2];

  if (d == 0) {
    const size_t bt = (size_t)b * Tz + t;
    const float bb = ba[bt * (2 * Hz) + h];
    const float aa = ba[bt * (2 * Hz) + Hz + h];
    const float Ax = __expf(fmaxf(A_log[h], -2.3f));
    const float x  = aa + dt_bias[h];
    const float sp = (x > 20.f) ? x : log1pf(__expf(x));
    beta[bt * Hz + h] = 1.f / (1.f + __expf(-bb));
    g[bt * Hz + h]    = -Ax * sp;
  }
}

// ---------------------------------------------------------------------------
// Gated delta-rule recurrence.  One 64-thread block per (b,h); thread j owns
// state column S[:,j] (64 f32 VGPRs).  k_t / q_t broadcast via LDS.
// ---------------------------------------------------------------------------
__global__ __launch_bounds__(64)
void gdn_recurrence(const float* __restrict__ qp, const float* __restrict__ kp,
                    const float* __restrict__ vp, const float* __restrict__ g,
                    const float* __restrict__ beta, float* __restrict__ y_raw) {
  const int bh = blockIdx.x;
  const int h = bh % Hz;
  const int b = bh / Hz;
  const int j = threadIdx.x;

  float S[HDz];
#pragma unroll
  for (int i = 0; i < HDz; ++i) S[i] = 0.f;

  __shared__ float kq[2 * HDz];

  for (int t = 0; t < Tz; ++t) {
    const size_t base = ((size_t)b * Tz + t) * KVz + h * HDz;
    const size_t gi   = ((size_t)b * Tz + t) * Hz + h;
    const float kj = kp[base + j];
    const float qj = qp[base + j];
    const float vj = vp[base + j];
    kq[j]       = kj;
    kq[HDz + j] = qj;
    const float gt = g[gi];
    const float bt = beta[gi];
    __syncthreads();

    const float decay = __expf(gt);
    float pred = 0.f;
#pragma unroll
    for (int kk = 0; kk < HDz; ++kk) {
      S[kk] *= decay;
      pred += kq[kk] * S[kk];
    }
    const float delta = (vj - pred) * bt;
    float o = 0.f;
#pragma unroll
    for (int kk = 0; kk < HDz; ++kk) {
      S[kk] += kq[kk] * delta;
      o += kq[HDz + kk] * S[kk];
    }
    y_raw[base + j] = o;
    __syncthreads();
  }
}

// ---------------------------------------------------------------------------
// Gated RMSNorm -> bf16 output for the out-projection GEMM.
// ---------------------------------------------------------------------------
__global__ __launch_bounds__(64)
void gdn_norm_gate(const float* __restrict__ y_raw,
                   const float* __restrict__ qkvz,
                   const float* __restrict__ norm_w, __bf16* __restrict__ y16) {
  const int bid = blockIdx.x;
  const int h = bid % Hz;
  const int t = (bid / Hz) % Tz;
  const int b = bid / (Hz * Tz);
  const int d = threadIdx.x;

  __shared__ float red[HDz];

  const size_t base = ((size_t)b * Tz + t) * KVz + h * HDz;
  const float o = y_raw[base + d];

  red[d] = o * o;
  __syncthreads();
  for (int off = 32; off > 0; off >>= 1) {
    if (d < off) red[d] += red[d + off];
    __syncthreads();
  }
  const float ms = red[0] * (1.f / HDz);

  const float z = qkvz[((size_t)b * Tz + t) * (4 * KVz) + 3 * KVz + h * HDz + d];
  y16[base + d] = (__bf16)(o * rsqrtf(ms + 1e-6f) * norm_w[d] * silu_f(z));
}

// ---------------------------------------------------------------------------
// Launch
// ---------------------------------------------------------------------------
extern "C" void kernel_launch(void* const* d_in, const int* in_sizes, int n_in,
                              void* d_out, int out_size, void* d_ws, size_t ws_size,
                              hipStream_t stream) {
  (void)in_sizes; (void)n_in; (void)out_size; (void)ws_size;

  const float* x       = (const float*)d_in[0];
  const float* W_in    = (const float*)d_in[1];
  const float* W_ba    = (const float*)d_in[2];
  const float* conv_w  = (const float*)d_in[3];
  const float* dt_bias = (const float*)d_in[4];
  const float* A_log   = (const float*)d_in[5];
  const float* norm_w  = (const float*)d_in[6];
  const float* W_o     = (const float*)d_in[7];
  float* out = (float*)d_out;

  const size_t BT = (size_t)Bz * Tz;                 // 8192
  char* w = (char*)d_ws;
  float* qkvz = (float*)w;  w += BT * 4 * KVz * sizeof(float);
  float* ba   = (float*)w;  w += BT * 2 * Hz  * sizeof(float);
  float* qp   = (float*)w;  w += BT * KVz     * sizeof(float);
  float* kp   = (float*)w;  w += BT * KVz     * sizeof(float);
  float* vp   = (float*)w;  w += BT * KVz     * sizeof(float);
  float* g    = (float*)w;  w += BT * Hz      * sizeof(float);
  float* beta = (float*)w;  w += BT * Hz      * sizeof(float);
  float* ybf  = (float*)w;  w += BT * KVz     * sizeof(float);
  __bf16* xb   = (__bf16*)w; w += BT * Dz          * sizeof(__bf16);
  __bf16* Winb = (__bf16*)w; w += (size_t)Dz * 4 * KVz * sizeof(__bf16);
  __bf16* Wob  = (__bf16*)w; w += (size_t)KVz * Dz * sizeof(__bf16);
  __bf16* y16  = (__bf16*)w; w += BT * KVz         * sizeof(__bf16);

  // 0) one-shot conversions to bf16
  gdn_cvt_bf16<<<(unsigned)(BT * Dz / 1024), 256, 0, stream>>>(x, xb);
  gdn_cvt_bf16<<<(unsigned)((size_t)Dz * 4 * KVz / 1024), 256, 0, stream>>>(W_in, Winb);
  gdn_cvt_bf16<<<(unsigned)((size_t)KVz * Dz / 1024), 256, 0, stream>>>(W_o, Wob);

  // 1) qkvz = x @ W_in  (8192 x 1024 x 4096)
  gdn_wmma_gemm_bf16<<<dim3(BT / 128, (4 * KVz) / 128), 128, 0, stream>>>(
      xb, Winb, qkvz, (int)BT, 4 * KVz, Dz);

  // 2) ba = x @ W_ba
  gdn_ba_gemm<<<(unsigned)(BT * 2 * Hz / 256), 256, 0, stream>>>(x, W_ba, ba);

  // 3) conv + SiLU + l2norm + gates
  gdn_conv_prep<<<(unsigned)(BT * Hz), 64, 0, stream>>>(
      qkvz, ba, conv_w, dt_bias, A_log, qp, kp, vp, g, beta);

  // 4) recurrence: one block per (b,h)
  gdn_recurrence<<<Bz * Hz, 64, 0, stream>>>(qp, kp, vp, g, beta, ybf);

  // 5) gated RMSNorm -> bf16 y
  gdn_norm_gate<<<(unsigned)(BT * Hz), 64, 0, stream>>>(ybf, qkvz, norm_w, y16);

  // 6) out = y @ W_o  (8192 x 1024 x 1024)
  gdn_wmma_gemm_bf16<<<dim3(BT / 128, Dz / 128), 128, 0, stream>>>(
      y16, Wob, out, (int)BT, Dz, KVz);
}
